// VoxelMlp_65446711656912
// MI455X (gfx1250) — matrix operand order repaired and checked by reading, not compiled, not run on hardware
//
#include <hip/hip_runtime.h>
#include <math.h>

typedef __attribute__((ext_vector_type(16))) _Float16 v16h;
typedef __attribute__((ext_vector_type(8)))  _Float16 v8h;
typedef __attribute__((ext_vector_type(8)))  float    v8f;

#define N_RAYS   4096
#define PPR      64
#define M_TOTAL  (N_RAYS*PPR)
#define GRID_N   128
#define VD       12
#define P2_G     64
#define P4_G     32

// Pre-shuffled f16 weight-fragment regions (offsets in halfs).
// Region size = Krounds*32*N  (same element count as padded matrix).
#define W_DTW0   0        // Kr=3, N=128 (87 valid K)   -> 12288
#define W_DTW1   12288    // Kr=4, N=128                -> 16384
#define W_FNW0   28672    // Kr=7, N=128 (207 valid K)  -> 28672
#define W_DECWF  57344    // Kr=4, N=128                -> 16384
#define W_DECW0  73728    // Kr=5, N=64  (149 valid K)  -> 10240
#define W_TOTAL  83968

__device__ __constant__ float kActShift = -4.5951198501345898f; // log(0.01/0.99)

// ---------------------------------------------------------------------------
// WMMA GEMM over an M-row activation tile in LDS (f16, 16B-aligned rows),
// weights read as pre-shuffled B fragments (v16h = 2x global_load_b128),
// A fragments as 2x ds_load_b128 + shuffle.  blockDim = 256 (8 wave32s).
// ---------------------------------------------------------------------------
__device__ __forceinline__ void gemm_layer(const _Float16* A, int lda,
                                           const _Float16* Wf,
                                           const float* bias, int Krounds,
                                           _Float16* C, int ldc,
                                           int Mrows, int Ncols, bool do_relu)
{
    const int lane = threadIdx.x & 31;
    const int wave = threadIdx.x >> 5;
    const int mT = Mrows >> 4, nT = Ncols >> 4, tot = mT * nT;
    for (int t = wave; t < tot; t += 8) {
        const int m0  = (t % mT) << 4;
        const int nTi = t / mT;
        v8f acc = {};
        const int am    = m0 + (lane & 15);
        const int abase = am * lda + ((lane >> 4) << 3);   // +8 K for hi half-wave
        const _Float16* bp = Wf + ((size_t)(nTi * Krounds) * 32 + lane) * 16;
        for (int kk = 0; kk < Krounds; ++kk) {
            const int k0 = kk << 5;
            v8h alo = *(const v8h*)(A + abase + k0);        // K = k0+base .. +7
            v8h ahi = *(const v8h*)(A + abase + k0 + 16);   // K = k0+base+16 .. +23
            v16h a = __builtin_shufflevector(alo, ahi,
                     0,1,2,3,4,5,6,7,8,9,10,11,12,13,14,15);
            v16h b = *(const v16h*)(bp + (size_t)kk * (32 * 16));
            acc = __builtin_amdgcn_wmma_f32_16x16x32_f16(false, a, false, b,
                                                         (short)0, acc, false, false);
        }
        const int   cn = (nTi << 4) + (lane & 15);
        const float bv = bias[cn];
        const int   cm = m0 + ((lane >> 4) << 3);
        #pragma unroll
        for (int v = 0; v < 8; ++v) {
            float x = acc[v] + bv;
            if (do_relu) x = fmaxf(x, 0.0f);
            C[(cm + v) * ldc + cn] = (_Float16)x;
        }
    }
}

// ---------------------------------------------------------------------------
// Trilinear interpolation helpers (align_corners, world coords in [-1,1])
// ---------------------------------------------------------------------------
__device__ __forceinline__ void trilerp_setup(int G, float px, float py, float pz,
                                              size_t* off, float* w)
{
    const float dims = (float)(G - 1);
    float tx = fminf(fmaxf((px + 1.0f) * 0.5f * dims, 0.0f), dims - 1e-4f);
    float ty = fminf(fmaxf((py + 1.0f) * 0.5f * dims, 0.0f), dims - 1e-4f);
    float tz = fminf(fmaxf((pz + 1.0f) * 0.5f * dims, 0.0f), dims - 1e-4f);
    int ix = (int)tx, iy = (int)ty, iz = (int)tz;
    float fx = tx - ix, fy = ty - iy, fz = tz - iz;
    size_t G2 = (size_t)G * G;
    size_t base = (size_t)ix * G2 + (size_t)iy * G + iz;
    int n = 0;
    for (int dx = 0; dx < 2; ++dx)
        for (int dy = 0; dy < 2; ++dy)
            for (int dz = 0; dz < 2; ++dz) {
                off[n] = base + (size_t)dx * G2 + (size_t)dy * G + dz;
                w[n] = (dx ? fx : 1.f - fx) * (dy ? fy : 1.f - fy) * (dz ? fz : 1.f - fz);
                ++n;
            }
}

__device__ __forceinline__ float trilerp1(const float* g, int G,
                                          float px, float py, float pz)
{
    size_t off[8]; float w[8];
    trilerp_setup(G, px, py, pz, off, w);
    float s = 0.f;
    #pragma unroll
    for (int k = 0; k < 8; ++k) s += g[off[k]] * w[k];
    return s;
}

__device__ __forceinline__ void trilerp12(const float* g, int G,
                                          float px, float py, float pz, float* out)
{
    size_t off[8]; float w[8];
    trilerp_setup(G, px, py, pz, off, w);
    const size_t G3 = (size_t)G * G * G;
    for (int c = 0; c < 12; ++c) {
        const float* gc = g + (size_t)c * G3;
        float s = 0.f;
        #pragma unroll
        for (int k = 0; k < 8; ++k) s += gc[off[k]] * w[k];
        out[c] = s;
    }
}

// ---------------------------------------------------------------------------
// K0: timenet  tf = relu(temb @ w0 + b0) @ w1 + b1     (1 block x 128 thr)
// ---------------------------------------------------------------------------
__global__ void timenet_kernel(const float* ft, const float* w0, const float* b0,
                               const float* w1, const float* b1, float* tf)
{
    __shared__ float h[128];
    const float t = ft[0];
    float te[9];
    te[0] = t;
    #pragma unroll
    for (int f = 0; f < 4; ++f) {
        te[1 + f] = sinf(t * (float)(1 << f));
        te[5 + f] = cosf(t * (float)(1 << f));
    }
    const int j = threadIdx.x;
    if (j < 128) {
        float acc = b0[j];
        #pragma unroll
        for (int i = 0; i < 9; ++i) acc += te[i] * w0[i * 128 + j];
        h[j] = fmaxf(acc, 0.0f);
    }
    __syncthreads();
    if (j < 60) {
        float acc = b1[j];
        for (int i = 0; i < 128; ++i) acc += h[i] * w1[i * 60 + j];
        tf[j] = acc;
    }
}

// ---------------------------------------------------------------------------
// K0b: per-ray view-direction embedding (21 floats / ray)
// ---------------------------------------------------------------------------
__global__ void vemb_kernel(const float* vd, float* vemb)
{
    int r = blockIdx.x * 256 + threadIdx.x;
    if (r >= N_RAYS) return;
    float v[3] = {vd[r*3+0], vd[r*3+1], vd[r*3+2]};
    float* o = vemb + (size_t)r * 21;
    o[0] = v[0]; o[1] = v[1]; o[2] = v[2];
    #pragma unroll
    for (int c = 0; c < 3; ++c)
        #pragma unroll
        for (int f = 0; f < 3; ++f) {
            float ang = v[c] * (float)(1 << f);
            o[3  + c*3 + f] = sinf(ang);
            o[12 + c*3 + f] = cosf(ang);
        }
}

// ---------------------------------------------------------------------------
// K1: 2x average-pool along all three spatial dims  (pool4 = pool2(pool2))
// ---------------------------------------------------------------------------
__global__ void poolhalf_kernel(const float* src, float* dst, int Gin)
{
    const int Go = Gin >> 1;
    const int Go3 = Go * Go * Go;
    int i = blockIdx.x * 256 + threadIdx.x;
    if (i >= VD * Go3) return;
    int c = i / Go3, rem = i % Go3;
    int ix = rem / (Go * Go), iy = (rem / Go) % Go, iz = rem % Go;
    const float* b = src + (size_t)c * Gin * Gin * Gin;
    float s = 0.f;
    #pragma unroll
    for (int dx = 0; dx < 2; ++dx)
        #pragma unroll
        for (int dy = 0; dy < 2; ++dy)
            #pragma unroll
            for (int dz = 0; dz < 2; ++dz)
                s += b[((size_t)(2*ix+dx) * Gin + (2*iy+dy)) * Gin + (2*iz+dz)];
    dst[i] = s * 0.125f;
}

// ---------------------------------------------------------------------------
// Kw: build f16 weights directly in WMMA B-fragment order:
//     halfs[((nT*Kr + kk)*32 + lane)*16 + e]  <-  w[k][n]
//     with  k = kk*32 + 16*(e>=8) + 8*(lane>=16) + (e&7),  n = nT*16 + (lane&15)
//     (K padded with zero rows to a multiple of 32)
// ---------------------------------------------------------------------------
__global__ void prep_w16_kernel(const float* dt_w0, const float* dt_w1,
                                const float* fn_w0, const float* dec_wf,
                                const float* dec_w0, _Float16* w16)
{
    int i = blockIdx.x * 256 + threadIdx.x;
    if (i >= W_TOTAL) return;
    const float* src; int Kr, N, Kv, j;
    if (i < W_DTW1)       { src = dt_w0;  Kr = 3; N = 128; Kv = 87;  j = i - W_DTW0; }
    else if (i < W_FNW0)  { src = dt_w1;  Kr = 4; N = 128; Kv = 128; j = i - W_DTW1; }
    else if (i < W_DECWF) { src = fn_w0;  Kr = 7; N = 128; Kv = 207; j = i - W_FNW0; }
    else if (i < W_DECW0) { src = dec_wf; Kr = 4; N = 128; Kv = 128; j = i - W_DECWF; }
    else                  { src = dec_w0; Kr = 5; N = 64;  Kv = 149; j = i - W_DECW0; }
    int e    = j & 15;
    int lane = (j >> 4) & 31;
    int frag = j >> 9;            // nT*Kr + kk
    int kk   = frag % Kr;
    int nT   = frag / Kr;
    int k = kk * 32 + ((e >> 3) << 4) + ((lane >> 4) << 3) + (e & 7);
    int n = (nT << 4) + (lane & 15);
    w16[i] = (_Float16)((k < Kv) ? src[k * N + n] : 0.f);
}

// ---------------------------------------------------------------------------
// K2: deformation net, fused per 64-point tile
//     h[64x96] -> relu GEMM -> relu GEMM -> 128x3 head -> pts_d
// ---------------------------------------------------------------------------
__global__ void deform_kernel(const float* ray_pts, const float* tf,
                              const _Float16* w16, const float* dt_b0,
                              const float* dt_b1, const float* dt_wo,
                              const float* dt_bo, float* ptsd)
{
    __shared__ alignas(16) _Float16 hbuf[64 * 96];
    __shared__ alignas(16) _Float16 a1[64 * 128];
    __shared__ alignas(16) _Float16 a2[64 * 128];
    const int g0 = blockIdx.x * 64;

    for (int i = threadIdx.x; i < 64 * 96; i += 256) {
        int p = i / 96, c = i % 96, g = g0 + p;
        float v;
        if (c < 3) {
            v = ray_pts[g * 3 + c];
        } else if (c < 27) {
            int j = c - 3;
            if (j < 12) { int cc = j >> 2, f = j & 3; v = sinf(ray_pts[g*3+cc] * (float)(1 << f)); }
            else { j -= 12; int cc = j >> 2, f = j & 3; v = cosf(ray_pts[g*3+cc] * (float)(1 << f)); }
        } else if (c < 87) {
            v = tf[c - 27];
        } else v = 0.0f;
        hbuf[i] = (_Float16)v;
    }
    __syncthreads();
    gemm_layer(hbuf, 96, w16 + W_DTW0, dt_b0, 3, a1, 128, 64, 128, true);
    __syncthreads();
    gemm_layer(a1, 128, w16 + W_DTW1, dt_b1, 4, a2, 128, 64, 128, true);
    __syncthreads();
    if (threadIdx.x < 192) {
        int p = threadIdx.x / 3, c = threadIdx.x % 3, g = g0 + p;
        float acc = dt_bo[c];
        for (int k = 0; k < 128; ++k) acc += (float)a2[p * 128 + k] * dt_wo[k * 3 + c];
        ptsd[g * 3 + c] = ray_pts[g * 3 + c] + acc;
    }
}

// ---------------------------------------------------------------------------
// K3: density trilerp -> softplus -> alpha
// ---------------------------------------------------------------------------
__global__ void alpha_kernel(const float* density, const float* ptsd, float* alpha)
{
    int g = blockIdx.x * 256 + threadIdx.x;
    if (g >= M_TOTAL) return;
    float d = trilerp1(density, GRID_N, ptsd[g*3], ptsd[g*3+1], ptsd[g*3+2]);
    float x = d + kActShift;
    float sp = (x > 20.f) ? x : log1pf(expf(x));
    alpha[g] = 1.0f - expf(-sp * 0.5f);
}

// ---------------------------------------------------------------------------
// K4: per-ray transmittance scan -> weights; init out with alphainv_last
// ---------------------------------------------------------------------------
__global__ void weights_kernel(const float* alpha, float* wgt, float* out)
{
    int r = blockIdx.x * 256 + threadIdx.x;
    if (r >= N_RAYS) return;
    float T = 1.0f;
    for (int j = 0; j < PPR; ++j) {
        float a = alpha[r * PPR + j];
        wgt[r * PPR + j] = a * T;
        T *= fmaxf(1.0f - a, 1e-10f);
    }
    out[r*3+0] = T; out[r*3+1] = T; out[r*3+2] = T;
}

// ---------------------------------------------------------------------------
// K5: feature trilerp x3 scales + pos-enc -> featurenet -> decoder -> RGB
//     -> weighted atomic composite into out   (32-point tiles)
// ---------------------------------------------------------------------------
__global__ void feat_dec_kernel(const float* feat, const float* pool2,
                                const float* pool4, const float* ptsd,
                                const _Float16* w16, const float* fn_b0,
                                const float* dec_bf, const float* dec_b0,
                                const float* dec_w1, const float* dec_b1,
                                const float* vemb, const float* wgt, float* out)
{
    __shared__ alignas(16) _Float16 xb[32 * 224];
    __shared__ alignas(16) _Float16 hf[32 * 128];
    __shared__ alignas(16) _Float16 yb[32 * 160];
    __shared__ alignas(16) _Float16 zb[32 * 64];
    const int g0 = blockIdx.x * 32;

    // Phase A: multi-scale trilerp into cols 0..35
    if (threadIdx.x < 96) {
        int p = threadIdx.x / 3, s = threadIdx.x % 3, g = g0 + p;
        float px = ptsd[g*3], py = ptsd[g*3+1], pz = ptsd[g*3+2];
        const float* gg = (s == 0) ? feat : (s == 1 ? pool2 : pool4);
        int G = (s == 0) ? GRID_N : (s == 1 ? P2_G : P4_G);
        float f12[12];
        trilerp12(gg, G, px, py, pz, f12);
        for (int ch = 0; ch < 12; ++ch)
            xb[p * 224 + s * 12 + ch] = (_Float16)f12[ch];
    }
    __syncthreads();
    // Phase B: positional encodings cols 36..223 + view-emb cols of y
    for (int i = threadIdx.x; i < 32 * 188; i += 256) {
        int p = i / 188, col = 36 + i % 188, g = g0 + p;
        float v;
        if (col < 108)       { int q = col - 36;  int c = q >> 1, f = q & 1; v = sinf((float)xb[p*224+c] * (float)(1 << f)); }
        else if (col < 180)  { int q = col - 108; int c = q >> 1, f = q & 1; v = cosf((float)xb[p*224+c] * (float)(1 << f)); }
        else if (col < 183)  { v = ptsd[g*3 + (col - 180)]; }
        else if (col < 195)  { int q = col - 183; int c = q >> 2, f = q & 3; v = sinf(ptsd[g*3+c] * (float)(1 << f)); }
        else if (col < 207)  { int q = col - 195; int c = q >> 2, f = q & 3; v = cosf(ptsd[g*3+c] * (float)(1 << f)); }
        else v = 0.0f;
        xb[p * 224 + col] = (_Float16)v;
    }
    for (int i = threadIdx.x; i < 32 * 32; i += 256) {
        int p = i >> 5, j = i & 31;
        float v = (j < 21) ? vemb[(size_t)((g0 + p) >> 6) * 21 + j] : 0.0f;
        yb[p * 160 + 128 + j] = (_Float16)v;
    }
    __syncthreads();
    gemm_layer(xb, 224, w16 + W_FNW0, fn_b0, 7, hf, 128, 32, 128, true);
    __syncthreads();
    gemm_layer(hf, 128, w16 + W_DECWF, dec_bf, 4, yb, 160, 32, 128, false);
    __syncthreads();
    gemm_layer(yb, 160, w16 + W_DECW0, dec_b0, 5, zb, 64, 32, 64, true);
    __syncthreads();
    if (threadIdx.x < 96) {
        int p = threadIdx.x / 3, c = threadIdx.x % 3, g = g0 + p;
        float acc = dec_b1[c];
        for (int k = 0; k < 64; ++k) acc += (float)zb[p * 64 + k] * dec_w1[k * 3 + c];
        float rgb = 1.0f / (1.0f + expf(-acc));
        atomicAdd(&out[(g >> 6) * 3 + c], wgt[g] * rgb);
    }
}

// ---------------------------------------------------------------------------
extern "C" void kernel_launch(void* const* d_in, const int* in_sizes, int n_in,
                              void* d_out, int out_size, void* d_ws, size_t ws_size,
                              hipStream_t stream)
{
    (void)in_sizes; (void)n_in; (void)out_size; (void)ws_size;
    const float* ray_pts    = (const float*)d_in[0];
    const float* viewdirs   = (const float*)d_in[1];
    const float* frame_time = (const float*)d_in[2];
    const float* feature    = (const float*)d_in[3];
    const float* density    = (const float*)d_in[4];
    const float* tn_w0 = (const float*)d_in[5];
    const float* tn_b0 = (const float*)d_in[6];
    const float* tn_w1 = (const float*)d_in[7];
    const float* tn_b1 = (const float*)d_in[8];
    const float* dt_w0 = (const float*)d_in[9];
    const float* dt_b0 = (const float*)d_in[10];
    const float* dt_w1 = (const float*)d_in[11];
    const float* dt_b1 = (const float*)d_in[12];
    const float* dt_wo = (const float*)d_in[13];
    const float* dt_bo = (const float*)d_in[14];
    const float* fn_w0 = (const float*)d_in[15];
    const float* fn_b0 = (const float*)d_in[16];
    const float* dec_wf = (const float*)d_in[17];
    const float* dec_bf = (const float*)d_in[18];
    const float* dec_w0 = (const float*)d_in[19];
    const float* dec_b0 = (const float*)d_in[20];
    const float* dec_w1 = (const float*)d_in[21];
    const float* dec_b1 = (const float*)d_in[22];
    // d_in[23] = ray_id: points are contiguous per ray, so ray = g >> 6.

    float* ws    = (float*)d_ws;
    float* tf    = ws;                                   // 64
    float* pool2 = tf    + 64;                           // 12*64^3
    float* pool4 = pool2 + (size_t)VD * P2_G*P2_G*P2_G;  // 12*32^3
    float* ptsd  = pool4 + (size_t)VD * P4_G*P4_G*P4_G;  // M*3
    float* wgt   = ptsd  + (size_t)M_TOTAL * 3;          // M
    float* alpha = wgt   + M_TOTAL;                      // M
    float* vembp = alpha + M_TOTAL;                      // 4096*21
    _Float16* w16 = (_Float16*)(vembp + (size_t)N_RAYS * 21);  // W_TOTAL halfs (32B-aligned)
    float* out = (float*)d_out;

    timenet_kernel<<<1, 128, 0, stream>>>(frame_time, tn_w0, tn_b0, tn_w1, tn_b1, tf);
    vemb_kernel<<<N_RAYS / 256, 256, 0, stream>>>(viewdirs, vembp);
    poolhalf_kernel<<<(VD * P2_G*P2_G*P2_G + 255) / 256, 256, 0, stream>>>(feature, pool2, GRID_N);
    poolhalf_kernel<<<(VD * P4_G*P4_G*P4_G + 255) / 256, 256, 0, stream>>>(pool2, pool4, P2_G);
    prep_w16_kernel<<<(W_TOTAL + 255) / 256, 256, 0, stream>>>(dt_w0, dt_w1, fn_w0, dec_wf, dec_w0, w16);
    deform_kernel<<<M_TOTAL / 64, 256, 0, stream>>>(ray_pts, tf, w16, dt_b0, dt_b1, dt_wo, dt_bo, ptsd);
    alpha_kernel<<<M_TOTAL / 256, 256, 0, stream>>>(density, ptsd, alpha);
    weights_kernel<<<N_RAYS / 256, 256, 0, stream>>>(alpha, wgt, out);
    feat_dec_kernel<<<M_TOTAL / 32, 256, 0, stream>>>(feature, pool2, pool4, ptsd, w16,
                                                      fn_b0, dec_bf, dec_b0, dec_w1, dec_b1,
                                                      vembp, wgt, out);
}
